// RPNHeadWraper_1202590843768
// MI455X (gfx1250) — matrix operand order repaired and checked by reading, not compiled, not run on hardware
//
#include <hip/hip_runtime.h>
#include <hip/hip_bf16.h>

typedef __attribute__((ext_vector_type(16))) _Float16 v16h;
typedef __attribute__((ext_vector_type(8)))  float    v8f;
typedef __attribute__((ext_vector_type(4)))  int      v4i;

union AFrag { v16h v; uint4 q[2]; };

// ---- CDNA5 async global->LDS copy (ASYNCcnt path), with sync fallback -----
#if defined(__gfx1250__) && __has_builtin(__builtin_amdgcn_global_load_async_to_lds_b128) && __has_builtin(__builtin_amdgcn_s_wait_asynccnt)
#define HAVE_ASYNC 1
#define ASYNC_CP16(dst, src)                                                    \
    __builtin_amdgcn_global_load_async_to_lds_b128(                             \
        (__attribute__((address_space(1))) v4i*)(unsigned long long)(src),      \
        (__attribute__((address_space(3))) v4i*)(unsigned long long)(dst),      \
        0, 0)
#define ASYNC_WAIT(n) __builtin_amdgcn_s_wait_asynccnt(n)
#else
#define HAVE_ASYNC 0
#define ASYNC_CP16(dst, src) (*(uint4*)(dst) = *(const uint4*)(src))
#define ASYNC_WAIT(n) ((void)0)
#endif

// ---------------------------------------------------------------------------
// Kernel 0: repack weights into WMMA-friendly f16 layouts.
//   w2  : per-K-chunk B^T tiles  [kc=72][n=256][kk=32]   (contiguous staging)
//   wcrT: fused cls(3)+reg(12)+pad head, transposed      [16][256]
// ---------------------------------------------------------------------------
__global__ __launch_bounds__(256)
void repack_weights(const float* __restrict__ wconv, const float* __restrict__ wcls,
                    const float* __restrict__ wreg, _Float16* __restrict__ w2,
                    _Float16* __restrict__ wcrT) {
    int k = blockIdx.x;      // 0..2303   k = tap*256 + c
    int n = threadIdx.x;     // 0..255
    int c = k & 255, tap = k >> 8;
    w2[((size_t)(k >> 5) * 256 + n) * 32 + (k & 31)] =
        (_Float16)wconv[((size_t)n * 256 + c) * 9 + tap];
    if (k < 16) {
        float v = (k < 3) ? wcls[k * 256 + n]
                          : ((k < 15) ? wreg[(k - 3) * 256 + n] : 0.0f);
        wcrT[k * 256 + n] = (_Float16)v;
    }
}

// ---------------------------------------------------------------------------
// Kernel P0: zero the 1-pixel border of the padded NHWC f16 buffer.
// ---------------------------------------------------------------------------
__global__ __launch_bounds__(256)
void pad_border(_Float16* __restrict__ pf, int H, int W) {
    const int Hp = H + 2, Wp = W + 2, P = Hp * Wp;
    const int tid = blockIdx.x * 256 + threadIdx.x;
    if (tid >= 2 * P) return;
    const int b = tid / P, pp = tid - b * P;
    const int py = pp / Wp, px = pp - py * Wp;
    if (py == 0 || py == Hp - 1 || px == 0 || px == Wp - 1) {
        uint4* dst = (uint4*)(pf + ((size_t)b * P + pp) * 256);
        const uint4 z = make_uint4(0, 0, 0, 0);
        #pragma unroll
        for (int i = 0; i < 32; ++i) dst[i] = z;
    }
}

// ---------------------------------------------------------------------------
// Kernel P1: NCHW f32 -> padded NHWC f16 (interior), tiled transpose via LDS.
//   grid: (ceil(HW/64), 8, B), block 256.
// ---------------------------------------------------------------------------
__global__ __launch_bounds__(256)
void nchw_to_pnhwc(const float* __restrict__ feat, _Float16* __restrict__ pf,
                   int H, int W) {
    __shared__ float tile[32][65];
    const int HW = H * W, Wp = W + 2;
    const int t = threadIdx.x;
    const int m0 = blockIdx.x * 64;
    const int cb = blockIdx.y * 32;
    const int b  = blockIdx.z;
    const float* fB = feat + ((size_t)b * 256 + cb) * HW;
    {
        const int x = t & 63, c0 = (t >> 6) * 8;
        #pragma unroll
        for (int i = 0; i < 8; ++i) {
            const int c = c0 + i, m = m0 + x;
            tile[c][x] = (m < HW) ? fB[(size_t)c * HW + m] : 0.0f;
        }
    }
    __syncthreads();
    {
        const int c = t & 31, p0 = t >> 5;
        _Float16* pfB = pf + (size_t)b * (H + 2) * Wp * 256;
        #pragma unroll
        for (int j = 0; j < 8; ++j) {
            const int p = p0 + j * 8, m = m0 + p;
            if (m < HW) {
                const int y = m / W, x = m - y * W;
                pfB[(((size_t)(y + 1)) * Wp + (x + 1)) * 256 + cb + c] =
                    (_Float16)tile[c][p];
            }
        }
    }
}

// ---------------------------------------------------------------------------
// Kernel 1: 3x3 conv 256->256 + bias + ReLU, implicit GEMM on WMMA.
//   Padded-NHWC f16 input -> branchless contiguous staging; double-buffered
//   async global->LDS pipeline (ASYNCcnt). grid: (ceil(HW/128), 2, B).
// ---------------------------------------------------------------------------
__global__ __launch_bounds__(256)
void conv3x3_wmma(const _Float16* __restrict__ pf, const float* __restrict__ bconv,
                  const _Float16* __restrict__ w2, _Float16* __restrict__ hout,
                  int H, int W) {
    const int HW = H * W, Wp = W + 2;
    __shared__ _Float16 lds_a[2][128 * 32];   // [buf][row][k]  2 x 8 KB
    __shared__ _Float16 lds_b[2][128 * 32];   // [buf][n][k]    2 x 8 KB

    const int t = threadIdx.x;
    const int wave = t >> 5, lane = t & 31;
    const int lane_lo = lane & 15, g = lane >> 4;
    const int mbase = blockIdx.x * 128;
    const int nbase = blockIdx.y * 128;
    const int b = blockIdx.z;

    // staging coords: 4 lanes per row, two row-blocks of 64
    const int q = t & 3, r0 = t >> 2;
    const int m0 = mbase + r0, m1 = mbase + r0 + 64;
    const int mc0 = (m0 < HW) ? m0 : (HW - 1);
    const int mc1 = (m1 < HW) ? m1 : (HW - 1);
    const int y0a = mc0 / W, x0a = mc0 - y0a * W;
    const int y1a = mc1 / W, x1a = mc1 - y1a * W;
    const _Float16* pfB = pf + (size_t)b * (H + 2) * Wp * 256;
    const size_t pix0 = (size_t)y0a * Wp + x0a;   // tap (dy=-1,dx=-1) base
    const size_t pix1 = (size_t)y1a * Wp + x1a;

    const int wm = wave & 3, wn = wave >> 2;
    const int Mblk = wm * 32, Nblk = wn * 64;

    v8f acc[2][4];
    #pragma unroll
    for (int mt = 0; mt < 2; ++mt)
        #pragma unroll
        for (int nt = 0; nt < 4; ++nt) {
            const float bias = bconv[nbase + Nblk + nt * 16 + lane_lo];
            #pragma unroll
            for (int r = 0; r < 8; ++r) acc[mt][nt][r] = bias;
        }

    auto stage = [&](int kc, int bufi) {
        const int tap = kc >> 3, cbase = (kc & 7) * 32;
        const int dy = tap / 3, dx = tap % 3;          // already +1 biased
        const size_t padd = (size_t)dy * Wp + dx;
        // A tile: two 16B transfers per thread
        ASYNC_CP16(&lds_a[bufi][r0 * 32 + q * 8],
                   pfB + (pix0 + padd) * 256 + cbase + q * 8);
        ASYNC_CP16(&lds_a[bufi][(r0 + 64) * 32 + q * 8],
                   pfB + (pix1 + padd) * 256 + cbase + q * 8);
        // B tile: contiguous 8 KB block, two 16B transfers per thread
        const _Float16* wsrc = w2 + ((size_t)kc * 256 + nbase) * 32 + t * 8;
        ASYNC_CP16(&lds_b[bufi][t * 8],        wsrc);
        ASYNC_CP16(&lds_b[bufi][t * 8 + 2048], wsrc + 2048);
    };

    stage(0, 0);
    for (int kc = 0; kc < 72; ++kc) {
        const int cur = kc & 1;
        if (kc + 1 < 72) {
            stage(kc + 1, cur ^ 1);
            ASYNC_WAIT(4);          // previous chunk complete, next in flight
        } else {
            ASYNC_WAIT(0);
        }
        __syncthreads();

        AFrag afrag[2];
        #pragma unroll
        for (int mt = 0; mt < 2; ++mt) {
            const int row = Mblk + mt * 16 + lane_lo;
            const uint4* p = (const uint4*)(&lds_a[cur][row * 32]);
            afrag[mt].q[0] = p[g];          // K = g*8 .. g*8+7
            afrag[mt].q[1] = p[2 + g];      // K = 16+g*8 ..
        }
        AFrag bfrag[4];
        #pragma unroll
        for (int nt = 0; nt < 4; ++nt) {
            const int col = Nblk + nt * 16 + lane_lo;
            const uint4* p = (const uint4*)(&lds_b[cur][col * 32]);
            bfrag[nt].q[0] = p[g];          // K = g*8 .. g*8+7
            bfrag[nt].q[1] = p[2 + g];
        }
        #pragma unroll
        for (int mt = 0; mt < 2; ++mt)
            #pragma unroll
            for (int nt = 0; nt < 4; ++nt)
                acc[mt][nt] = __builtin_amdgcn_wmma_f32_16x16x32_f16(
                    false, afrag[mt].v, false, bfrag[nt].v,
                    (short)0, acc[mt][nt], false, false);
        __syncthreads();
    }

    _Float16* hB = hout + (size_t)b * HW * 256;
    #pragma unroll
    for (int mt = 0; mt < 2; ++mt)
        #pragma unroll
        for (int r = 0; r < 8; ++r) {
            const int mm = mbase + Mblk + mt * 16 + r + 8 * g;
            if (mm < HW) {
                #pragma unroll
                for (int nt = 0; nt < 4; ++nt) {
                    const int nn = nbase + Nblk + nt * 16 + lane_lo;
                    const float v = acc[mt][nt][r];
                    hB[(size_t)mm * 256 + nn] = (_Float16)(v > 0.0f ? v : 0.0f);
                }
            }
        }
}

// ---------------------------------------------------------------------------
// Kernel 2: fused 1x1 heads (cls 3 + reg 12 + pad) as one N=16 WMMA tile.
//   grid: (ceil(HW/128), B), block 256. Async-staged A from h. K=256.
// ---------------------------------------------------------------------------
__global__ __launch_bounds__(256)
void conv1x1_wmma(const _Float16* __restrict__ h, const _Float16* __restrict__ wcrT,
                  const float* __restrict__ bcls, const float* __restrict__ breg,
                  float* __restrict__ outcr, int HW) {
    __shared__ _Float16 lds_a[128 * 32];
    __shared__ _Float16 lds_b[16 * 256];

    const int t = threadIdx.x;
    const int wave = t >> 5, lane = t & 31;
    const int lane_lo = lane & 15, g = lane >> 4;
    const int mbase = blockIdx.x * 128;
    const int bb = blockIdx.y;
    const _Float16* hB = h + (size_t)bb * HW * 256;

    // stage full head weights once (8 KB)
    ASYNC_CP16(lds_b + t * 16,     wcrT + t * 16);
    ASYNC_CP16(lds_b + t * 16 + 8, wcrT + t * 16 + 8);

    float bias;
    { const int n = lane_lo; bias = (n < 3) ? bcls[n] : ((n < 15) ? breg[n - 3] : 0.0f); }
    v8f acc;
    #pragma unroll
    for (int r = 0; r < 8; ++r) acc[r] = bias;

    const int q = t & 3, r0 = t >> 2;
    const int ma0 = mbase + r0,      mb0 = mbase + r0 + 64;
    const int mc0 = (ma0 < HW) ? ma0 : (HW - 1);
    const int mc1 = (mb0 < HW) ? mb0 : (HW - 1);

    for (int kc = 0; kc < 8; ++kc) {
        ASYNC_CP16(&lds_a[r0 * 32 + q * 8],
                   hB + (size_t)mc0 * 256 + kc * 32 + q * 8);
        ASYNC_CP16(&lds_a[(r0 + 64) * 32 + q * 8],
                   hB + (size_t)mc1 * 256 + kc * 32 + q * 8);
        ASYNC_WAIT(0);
        __syncthreads();

        AFrag a, bf;
        const int row = wave * 16 + lane_lo;
        const uint4* pa = (const uint4*)(lds_a + row * 32);
        a.q[0] = pa[g]; a.q[1] = pa[2 + g];
        const uint4* pb = (const uint4*)(lds_b + lane_lo * 256 + kc * 32);
        bf.q[0] = pb[g]; bf.q[1] = pb[2 + g];
        acc = __builtin_amdgcn_wmma_f32_16x16x32_f16(
            false, a.v, false, bf.v, (short)0, acc, false, false);
        __syncthreads();
    }

    float* outB = outcr + (size_t)bb * HW * 16;
    #pragma unroll
    for (int r = 0; r < 8; ++r) {
        const int mm = mbase + wave * 16 + r + 8 * g;
        if (mm < HW) outB[(size_t)mm * 16 + lane_lo] = acc[r];
    }
}

// ---------------------------------------------------------------------------
// Kernel 3: sigmoid + anchors + delta decode + clip.
// ---------------------------------------------------------------------------
__global__ __launch_bounds__(256)
void decode_kernel(const float* __restrict__ outcr, float* __restrict__ scores,
                   float* __restrict__ boxes) {
    const int Ktot = 268569;
    const int tid = blockIdx.x * blockDim.x + threadIdx.x;
    if (tid >= 2 * Ktot) return;
    const int b = tid / Ktot;
    const int j = tid - b * Ktot;

    const int   Koff[6]  = {0, 201600, 252000, 264600, 267750, 268569};
    const int   HWl[5]   = {67200, 16800, 4200, 1050, 273};
    const int   Wl[5]    = {336, 168, 84, 42, 21};
    const float strd[5]  = {4.f, 8.f, 16.f, 32.f, 64.f};
    const long  croff[5] = {0, 2150400, 2688000, 2822400, 2856000};

    int l = 0;
    while (l < 4 && j >= Koff[l + 1]) ++l;
    const int jl = j - Koff[l];
    const int m = jl / 3, a = jl - 3 * m;

    const float* cr = outcr + croff[l] + ((size_t)b * HWl[l] + m) * 16;
    const float score = 1.0f / (1.0f + expf(-cr[a]));
    const float MAXD = 4.135166556742356f;  // log(1000/16)
    const float dxv = cr[3 + 4 * a + 0];
    const float dyv = cr[3 + 4 * a + 1];
    const float dw  = fminf(cr[3 + 4 * a + 2], MAXD);
    const float dh  = fminf(cr[3 + 4 * a + 3], MAXD);

    const int W = Wl[l];
    const int x = m % W, y = m / W;
    const float s = strd[l];
    const float hr = (a == 0) ? 0.70710678118654752f : ((a == 1) ? 1.0f : 1.41421356237309515f);
    const float wr = (a == 0) ? 1.41421356237309515f : ((a == 1) ? 1.0f : 0.70710678118654752f);
    const float aw = s * 8.0f * wr, ah = s * 8.0f * hr;
    const float acx = x * s, acy = y * s;
    const float cx = acx + dxv * aw, cy = acy + dyv * ah;
    const float w = aw * expf(dw), hh = ah * expf(dh);
    const float x1 = fminf(fmaxf(cx - 0.5f * w, 0.0f), 1344.0f);
    const float y1 = fminf(fmaxf(cy - 0.5f * hh, 0.0f), 800.0f);
    const float x2 = fminf(fmaxf(cx + 0.5f * w, 0.0f), 1344.0f);
    const float y2 = fminf(fmaxf(cy + 0.5f * hh, 0.0f), 800.0f);

    scores[(size_t)b * Ktot + j] = score;
    float* bx = boxes + ((size_t)b * Ktot + j) * 4;
    bx[0] = x1; bx[1] = y1; bx[2] = x2; bx[3] = y2;
}

// ---------------------------------------------------------------------------
// Kernel 4: per-(level,batch) top-1000 selection.
// ---------------------------------------------------------------------------
__global__ __launch_bounds__(1024)
void topk_kernel(float* __restrict__ scores, const float* __restrict__ boxes,
                 float* __restrict__ cand) {
    __shared__ float sVal[1024]; __shared__ int sArg[1024];
    __shared__ float rVal[1024]; __shared__ int rArg[1024];
    __shared__ int winJ;
    const int Ktot = 268569;
    const int Koff[6] = {0, 201600, 252000, 264600, 267750, 268569};
    const int l = blockIdx.x, b = blockIdx.y, t = threadIdx.x;
    const int base = b * Ktot + Koff[l];
    const int Kl = Koff[l + 1] - Koff[l];

    float best = -1.0f; int bi = -1;
    for (int j = t; j < Kl; j += 1024) {
        const float v = scores[base + j];
        if (v > best) { best = v; bi = j; }
    }
    sVal[t] = best; sArg[t] = bi;
    __syncthreads();

    for (int k = 0; k < 1000; ++k) {
        rVal[t] = sVal[t]; rArg[t] = sArg[t];
        __syncthreads();
        for (int s = 512; s > 0; s >>= 1) {
            if (t < s && rVal[t + s] > rVal[t]) { rVal[t] = rVal[t + s]; rArg[t] = rArg[t + s]; }
            __syncthreads();
        }
        if (t == 0) {
            const float val = rVal[0]; const int j = rArg[0];
            float* co = cand + ((size_t)(b * 5 + l) * 1000 + k) * 5;
            if (val > 0.0f && j >= 0) {
                co[0] = val;
                const float* bx = boxes + ((size_t)base + j) * 4;
                co[1] = bx[0]; co[2] = bx[1]; co[3] = bx[2]; co[4] = bx[3];
                winJ = j;
            } else {
                co[0] = 0.f; co[1] = 0.f; co[2] = 0.f; co[3] = 0.f; co[4] = 0.f;
                winJ = -1;
            }
        }
        __syncthreads();
        const int wj = winJ;
        if (wj >= 0 && t == (wj & 1023)) {
            scores[base + wj] = -1.0f;
            float nb = -1.0f; int ni = -1;
            for (int j = t; j < Kl; j += 1024) {
                const float v = scores[base + j];
                if (v > nb) { nb = v; ni = j; }
            }
            sVal[t] = nb; sArg[t] = ni;
        }
        __syncthreads();
    }
}

// ---------------------------------------------------------------------------
// Kernel 5: greedy NMS (bitonic sort in LDS + serial suppression).
// ---------------------------------------------------------------------------
__global__ __launch_bounds__(256)
void nms_kernel(const float* __restrict__ cand, float* __restrict__ out) {
    __shared__ float s[8192];
    __shared__ short si[8192];
    __shared__ float bx[4];
    __shared__ int alive;
    __shared__ int scnt;
    const int b = blockIdx.x, t = threadIdx.x;
    const float* c = cand + (size_t)b * 5000 * 5;

    for (int j = t; j < 8192; j += 256) {
        if (j < 5000) { s[j] = c[j * 5]; si[j] = (short)j; }
        else          { s[j] = -2.0f;    si[j] = -1; }
    }
    __syncthreads();

    for (int ksz = 2; ksz <= 8192; ksz <<= 1)
        for (int jsz = ksz >> 1; jsz > 0; jsz >>= 1) {
            for (int i = t; i < 8192; i += 256) {
                const int ixj = i ^ jsz;
                if (ixj > i) {
                    const bool up = ((i & ksz) == 0);
                    const float a0 = s[i], a1 = s[ixj];
                    const bool doswap = up ? (a0 < a1) : (a0 > a1);
                    if (doswap) {
                        s[i] = a1; s[ixj] = a0;
                        const short t0 = si[i]; si[i] = si[ixj]; si[ixj] = t0;
                    }
                }
            }
            __syncthreads();
        }

    for (int i = 0; i < 5000; ++i) {
        if (t == 0) {
            alive = (s[i] >= 0.0f);
            if (alive) {
                const float* p = c + (size_t)si[i] * 5 + 1;
                bx[0] = p[0]; bx[1] = p[1]; bx[2] = p[2]; bx[3] = p[3];
            }
        }
        __syncthreads();
        if (alive) {
            const float ax1 = bx[0], ay1 = bx[1], ax2 = bx[2], ay2 = bx[3];
            const float areaA = (ax2 - ax1) * (ay2 - ay1);
            for (int j = i + 1 + t; j < 5000; j += 256) {
                if (s[j] >= 0.0f) {
                    const float* p = c + (size_t)si[j] * 5 + 1;
                    const float bx1 = p[0], by1 = p[1], bx2 = p[2], by2 = p[3];
                    const float iw = fmaxf(fminf(ax2, bx2) - fmaxf(ax1, bx1), 0.0f);
                    const float ih = fmaxf(fminf(ay2, by2) - fmaxf(ay1, by1), 0.0f);
                    const float inter = iw * ih;
                    const float areaB = (bx2 - bx1) * (by2 - by1);
                    const float iou = inter / (areaA + areaB - inter + 1e-6f);
                    if (iou > 0.7f) s[j] = -1.0f;
                }
            }
        }
        __syncthreads();
    }

    if (t == 0) {
        int cnt = 0;
        for (int i = 0; i < 5000 && cnt < 1000; ++i)
            if (s[i] >= 0.0f) {
                const float* p = c + (size_t)si[i] * 5 + 1;
                float* o = out + ((size_t)b * 1000 + cnt) * 4;
                o[0] = p[0]; o[1] = p[1]; o[2] = p[2]; o[3] = p[3];
                ++cnt;
            }
        scnt = cnt;
    }
    __syncthreads();
    for (int idx = scnt * 4 + t; idx < 4000; idx += 256)
        out[(size_t)b * 4000 + idx] = 0.0f;
}

// ---------------------------------------------------------------------------
extern "C" void kernel_launch(void* const* d_in, const int* in_sizes, int n_in,
                              void* d_out, int out_size, void* d_ws, size_t ws_size,
                              hipStream_t stream) {
    const float* feats[5] = {(const float*)d_in[0], (const float*)d_in[1],
                             (const float*)d_in[2], (const float*)d_in[3],
                             (const float*)d_in[4]};
    const float* wconv = (const float*)d_in[6];
    const float* bconv = (const float*)d_in[7];
    const float* wcls  = (const float*)d_in[8];
    const float* bcls  = (const float*)d_in[9];
    const float* wreg  = (const float*)d_in[10];
    const float* breg  = (const float*)d_in[11];

    static const int Hs[5]  = {200, 100, 50, 25, 13};
    static const int Ws_[5] = {336, 168, 84, 42, 21};
    const int B = 2, Ktot = 268569, sumHW = 89523;
    size_t sumPad = 0;
    for (int l = 0; l < 5; ++l) sumPad += (size_t)(Hs[l] + 2) * (Ws_[l] + 2);

    char* ws = (char*)d_ws;
    size_t off = 0;
    auto alloc = [&](size_t bytes) {
        size_t o = off; off = (off + bytes + 255) & ~(size_t)255; return o;
    };
    _Float16* w2    = (_Float16*)(ws + alloc((size_t)2304 * 256 * 2));
    _Float16* wcrT  = (_Float16*)(ws + alloc((size_t)16 * 256 * 2));
    _Float16* pfbuf = (_Float16*)(ws + alloc((size_t)B * sumPad * 256 * 2));
    _Float16* hbuf  = (_Float16*)(ws + alloc((size_t)B * sumHW * 256 * 2));
    float*    crbuf = (float*)   (ws + alloc((size_t)B * sumHW * 16 * 4));
    float*    scor  = (float*)   (ws + alloc((size_t)B * Ktot * 4));
    float*    boxes = (float*)   (ws + alloc((size_t)B * Ktot * 4 * 4));
    float*    cand  = (float*)   (ws + alloc((size_t)B * 5000 * 5 * 4));
    (void)ws_size; (void)in_sizes; (void)n_in; (void)out_size;

    repack_weights<<<2304, 256, 0, stream>>>(wconv, wcls, wreg, w2, wcrT);

    size_t pfoff = 0, hoff = 0, croff = 0;
    for (int l = 0; l < 5; ++l) {
        const int H = Hs[l], W = Ws_[l], HW = H * W;
        const int P = (H + 2) * (W + 2);
        pad_border<<<(2 * P + 255) / 256, 256, 0, stream>>>(pfbuf + pfoff, H, W);
        nchw_to_pnhwc<<<dim3((HW + 63) / 64, 8, B), 256, 0, stream>>>(
            feats[l], pfbuf + pfoff, H, W);
        conv3x3_wmma<<<dim3((HW + 127) / 128, 2, B), 256, 0, stream>>>(
            pfbuf + pfoff, bconv, w2, hbuf + hoff, H, W);
        conv1x1_wmma<<<dim3((HW + 127) / 128, B), 256, 0, stream>>>(
            hbuf + hoff, wcrT, bcls, breg, crbuf + croff, HW);
        pfoff += (size_t)B * P * 256;
        hoff  += (size_t)B * HW * 256;
        croff += (size_t)B * HW * 16;
    }

    decode_kernel<<<(2 * Ktot + 255) / 256, 256, 0, stream>>>(crbuf, scor, boxes);
    topk_kernel<<<dim3(5, B), 1024, 0, stream>>>(scor, boxes, cand);
    nms_kernel<<<B, 256, 0, stream>>>(cand, (float*)d_out);
}
